// MultiHeadAttention_50835232916148
// MI455X (gfx1250) — compile-verified, matrix operand order checked by reading, not run on hardware
//
#include <hip/hip_runtime.h>

// ---------------------------------------------------------------------------
// MI455X (gfx1250) multi-head attention block, bf16 WMMA pipeline.
// GEMMs: v_wmma_f32_16x16x32_bf16 + TDM (tensor_load_to_lds) double-buffered
// LDS staging. Flash attention: register-resident online softmax.
// ---------------------------------------------------------------------------

typedef __bf16 bf16;
typedef __attribute__((ext_vector_type(16))) __bf16 v16bf;
typedef __attribute__((ext_vector_type(8)))  __bf16 v8bf;
typedef __attribute__((ext_vector_type(4)))  __bf16 v4bf;
typedef __attribute__((ext_vector_type(8)))  float  v8f;
typedef __attribute__((ext_vector_type(4)))  unsigned int v4u;
typedef __attribute__((ext_vector_type(8)))  int v8i;
typedef __attribute__((ext_vector_type(4)))  int v4i;

#define HS 1024
#define NHEAD 16
#define HDIM 64
#define SL 1024
#define BS 8
#define NROWS (BS * SL)        // 8192
#define MASK_NEG -100000000.0f
#define LDST 72                // padded LDS row stride (elems); 144B

#if __has_builtin(__builtin_amdgcn_tensor_load_to_lds) && \
    __has_builtin(__builtin_amdgcn_s_wait_tensorcnt)
#define USE_TDM 1
#else
#define USE_TDM 0
#endif

__device__ __forceinline__ v8f wmma_bf16(v16bf a, v16bf b, v8f c) {
    return __builtin_amdgcn_wmma_f32_16x16x32_bf16(
        /*neg_a=*/false, a, /*neg_b=*/false, b,
        /*c_mod=*/(short)0, c, /*reuse_a=*/false, /*reuse_b=*/false);
}

// Load one 16x32 bf16 fragment (A layout; B mirrors it from a row-major
// [16 rows x K] source). Lanes 0-15 -> row = lane, kb = 0; lanes 16-31 ->
// row = lane-16, kb = 8; elems {kb..kb+7, kb+16..kb+23}.
__device__ __forceinline__ v16bf load_frag(const bf16* __restrict__ base,
                                           int stride, int lane) {
    int r  = lane & 15;
    int kb = (lane & 16) ? 8 : 0;
    const bf16* p = base + (size_t)r * stride + kb;
    v8bf lo = *(const v8bf*)(p);
    v8bf hi = *(const v8bf*)(p + 16);
    v16bf f;
#pragma unroll
    for (int i = 0; i < 8; ++i) { f[i] = lo[i]; f[8 + i] = hi[i]; }
    return f;
}

#if USE_TDM
// Tensor DMA: load a 128-row x 64-elem bf16 tile (row stride 1024 elems in
// global) into LDS at lds_addr, padding 4 dwords after each 32-dword row so
// the LDS row stride becomes LDST=72 elems. D# per CDNA5 ISA ch.8.
__device__ __forceinline__ void tdm_load_tile(unsigned lds_addr,
                                              const bf16* gptr) {
    unsigned long long ga = (unsigned long long)gptr;
    v4u g0;
    g0[0] = 1u;                                   // count=1 (valid user D#)
    g0[1] = lds_addr;                             // lds_addr (bytes)
    g0[2] = (unsigned)(ga & 0xFFFFFFFFu);         // global_addr[31:0]
    g0[3] = (unsigned)((ga >> 32) & 0x01FFFFFFu)  // global_addr[56:32]
          | (2u << 30);                           // type = 2 ("image")
    v8i g1;
    g1[0] = (int)((1u << 16)      // data_size = 2 bytes
                | (1u << 20)      // pad_enable
                | (4u << 22)      // pad_interval: 32 dwords (one 128B row)
                | (3u << 25));    // pad_amount: 4 dwords (16B -> stride 144B)
    // tensor_dim0 = tensor_dim1 = 1<<20 (never OOB); tile 64 x 128;
    // tensor_dim0_stride = 1024 elems.
    g1[1] = 0;                               // abar=0, tensor_dim0[15:0]=0
    g1[2] = 0x10;                            // tensor_dim0[31:16]=0x10, td1 lo=0
    g1[3] = (int)(0x10u | (64u << 16));      // tensor_dim1[31:16], tile_dim0=64
    g1[4] = 128;                             // tile_dim1 = 128, tile_dim2 = 0
    g1[5] = 1024;                            // tensor_dim0_stride[31:0]
    g1[6] = 0;
    g1[7] = 0;
    v4i gz4 = {0, 0, 0, 0};
    v8i gz8 = {0, 0, 0, 0, 0, 0, 0, 0};
    __builtin_amdgcn_tensor_load_to_lds(g0, g1, gz4, gz4, gz8, 0);
}
#endif

// ---------------------------------------------------------------------------
// Kernel 1: fp32 -> bf16 conversion (weights)
// ---------------------------------------------------------------------------
__global__ void cvt_bf16_kernel(const float* __restrict__ src,
                                bf16* __restrict__ dst, int n4) {
    int i = blockIdx.x * blockDim.x + threadIdx.x;
    int stride = gridDim.x * blockDim.x;
    for (; i < n4; i += stride) {
        float4 v = ((const float4*)src)[i];
        v4bf o = { (bf16)v.x, (bf16)v.y, (bf16)v.z, (bf16)v.w };
        ((v4bf*)dst)[i] = o;
    }
}

// ---------------------------------------------------------------------------
// Kernel 2: LayerNorm (1024), fp32 in -> bf16 out. One block per row.
// ---------------------------------------------------------------------------
__global__ __launch_bounds__(256) void ln_kernel(const float* __restrict__ x,
                                                 const float* __restrict__ w,
                                                 const float* __restrict__ b,
                                                 bf16* __restrict__ y) {
    int row  = blockIdx.x;
    int tid  = threadIdx.x;
    int lane = tid & 31, wave = tid >> 5;
    float4 v = ((const float4*)(x + (size_t)row * HS))[tid];
    float s  = v.x + v.y + v.z + v.w;
    float ss = v.x * v.x + v.y * v.y + v.z * v.z + v.w * v.w;
#pragma unroll
    for (int d = 16; d >= 1; d >>= 1) {
        s  += __shfl_xor(s,  d, 32);
        ss += __shfl_xor(ss, d, 32);
    }
    __shared__ float rs[8], rss[8];
    if (lane == 0) { rs[wave] = s; rss[wave] = ss; }
    __syncthreads();
    float tot = 0.f, tot2 = 0.f;
#pragma unroll
    for (int i = 0; i < 8; ++i) { tot += rs[i]; tot2 += rss[i]; }
    float mean = tot * (1.0f / HS);
    float var  = tot2 * (1.0f / HS) - mean * mean;
    float inv  = rsqrtf(var + 1e-5f);
    float4 wv  = ((const float4*)w)[tid];
    float4 bv  = ((const float4*)b)[tid];
    v4bf o = { (bf16)((v.x - mean) * inv * wv.x + bv.x),
               (bf16)((v.y - mean) * inv * wv.y + bv.y),
               (bf16)((v.z - mean) * inv * wv.z + bv.z),
               (bf16)((v.w - mean) * inv * wv.w + bv.w) };
    ((v4bf*)(y + (size_t)row * HS))[tid] = o;
}

// ---------------------------------------------------------------------------
// Shared GEMM mainloop: acc[2][4] = A(128x1024 tile @ m0) * B^T(@ n0).
// Block 128x128, 8 waves each 32x64.  TDM path: double-buffered LDS tiles,
// wave 0 issues DMA for step i+1 while all waves run WMMAs on step i.
// ---------------------------------------------------------------------------
__device__ __forceinline__ void gemm_mainloop(
        const bf16* __restrict__ A, const bf16* __restrict__ B,
        bf16* As /*[2][128*LDST]*/, bf16* Bs /*[2][128*LDST]*/,
        int m0, int n0, int tid, int lane, int wr, int wc, v8f acc[2][4]) {
#if USE_TDM
    int wave = tid >> 5;
    unsigned as_lds = (unsigned)(unsigned long long)As;
    unsigned bs_lds = (unsigned)(unsigned long long)Bs;
    const unsigned bufbytes = 128 * LDST * sizeof(bf16);
    if (wave == 0) {
        tdm_load_tile(as_lds, A + (size_t)m0 * HS);
        tdm_load_tile(bs_lds, B + (size_t)n0 * HS);
        __builtin_amdgcn_s_wait_tensorcnt(0);
    }
    __syncthreads();
    for (int k0 = 0; k0 < HS; k0 += 64) {
        int cur = (k0 >> 6) & 1;
        int nxt = cur ^ 1;
        if (wave == 0 && (k0 + 64) < HS) {
            tdm_load_tile(as_lds + nxt * bufbytes,
                          A + (size_t)m0 * HS + (k0 + 64));
            tdm_load_tile(bs_lds + nxt * bufbytes,
                          B + (size_t)n0 * HS + (k0 + 64));
        }
        const bf16* Ac = As + cur * (128 * LDST);
        const bf16* Bc = Bs + cur * (128 * LDST);
#pragma unroll
        for (int kk = 0; kk < 64; kk += 32) {
            v16bf a[2], b[4];
#pragma unroll
            for (int i = 0; i < 2; ++i)
                a[i] = load_frag(Ac + (wr * 32 + i * 16) * LDST + kk, LDST, lane);
#pragma unroll
            for (int j = 0; j < 4; ++j)
                b[j] = load_frag(Bc + (wc * 64 + j * 16) * LDST + kk, LDST, lane);
#pragma unroll
            for (int i = 0; i < 2; ++i)
#pragma unroll
                for (int j = 0; j < 4; ++j)
                    acc[i][j] = wmma_bf16(a[i], b[j], acc[i][j]);
        }
        if (wave == 0) __builtin_amdgcn_s_wait_tensorcnt(0);
        __syncthreads();
    }
#else
    for (int k0 = 0; k0 < HS; k0 += 64) {
#pragma unroll
        for (int j = 0; j < 4; ++j) {
            int c  = tid + 256 * j;
            int r  = c >> 3;
            int c8 = (c & 7) * 8;
            *(v8bf*)(As + r * LDST + c8) =
                *(const v8bf*)(A + (size_t)(m0 + r) * HS + k0 + c8);
            *(v8bf*)(Bs + r * LDST + c8) =
                *(const v8bf*)(B + (size_t)(n0 + r) * HS + k0 + c8);
        }
        __syncthreads();
#pragma unroll
        for (int kk = 0; kk < 64; kk += 32) {
            v16bf a[2], b[4];
#pragma unroll
            for (int i = 0; i < 2; ++i)
                a[i] = load_frag(As + (wr * 32 + i * 16) * LDST + kk, LDST, lane);
#pragma unroll
            for (int j = 0; j < 4; ++j)
                b[j] = load_frag(Bs + (wc * 64 + j * 16) * LDST + kk, LDST, lane);
#pragma unroll
            for (int i = 0; i < 2; ++i)
#pragma unroll
                for (int j = 0; j < 4; ++j)
                    acc[i][j] = wmma_bf16(a[i], b[j], acc[i][j]);
        }
        __syncthreads();
    }
#endif
}

// ---------------------------------------------------------------------------
// Kernel 3: QKV GEMM.  C[m,n] = sum_k Y[m,k]*Wqkv[n,k] + bias[n]
//   M=8192, N=3072 (t*1024+e), K=1024.
// Epilogue scatters Q[b,h,s,d] (x 1/8), K[b,h,s,d], Vt[b,h,d,s] in bf16.
// ---------------------------------------------------------------------------
__global__ __launch_bounds__(256) void qkv_gemm_kernel(
        const bf16* __restrict__ Y, const bf16* __restrict__ W,
        const float* __restrict__ bias,
        bf16* __restrict__ Q, bf16* __restrict__ K, bf16* __restrict__ Vt) {
    const int m0 = blockIdx.y * 128;
    const int n0 = blockIdx.x * 128;
#if USE_TDM
    __shared__ alignas(16) bf16 As[2][128 * LDST];
    __shared__ alignas(16) bf16 Bs[2][128 * LDST];
#else
    __shared__ alignas(16) bf16 As[1][128 * LDST];
    __shared__ alignas(16) bf16 Bs[1][128 * LDST];
#endif
    int tid = threadIdx.x, lane = tid & 31, wave = tid >> 5;
    int wr = wave >> 1, wc = wave & 1;
    v8f acc[2][4];
#pragma unroll
    for (int i = 0; i < 2; ++i)
#pragma unroll
        for (int j = 0; j < 4; ++j) acc[i][j] = (v8f)0.f;

    gemm_mainloop(Y, W, &As[0][0], &Bs[0][0], m0, n0, tid, lane, wr, wc, acc);

    int col   = lane & 15;
    int rbase = (lane & 16) ? 8 : 0;
#pragma unroll
    for (int i = 0; i < 2; ++i)
#pragma unroll
        for (int j = 0; j < 4; ++j) {
            int n = n0 + wc * 64 + j * 16 + col;
            int t = n >> 10, e = n & 1023;
            int h = e >> 6,  d = e & 63;
            float bval = bias[n];
#pragma unroll
            for (int ii = 0; ii < 8; ++ii) {
                int m  = m0 + wr * 32 + i * 16 + rbase + ii;
                int bi = m >> 10, s = m & 1023;
                float val = acc[i][j][ii] + bval;
                if (t == 0) {
                    Q[((size_t)(bi * NHEAD + h) * SL + s) * HDIM + d] =
                        (bf16)(val * 0.125f);   // fold 1/sqrt(64)
                } else if (t == 1) {
                    K[((size_t)(bi * NHEAD + h) * SL + s) * HDIM + d] = (bf16)val;
                } else {
                    Vt[((size_t)(bi * NHEAD + h) * HDIM + d) * SL + s] = (bf16)val;
                }
            }
        }
}

// ---------------------------------------------------------------------------
// Kernel 4: flash attention. Grid (8 qtiles, 16 heads, 8 batch).
// 8 waves/block; each wave owns 16 query rows, streams 32-key chunks with
// online softmax; per-wave LDS buffer transposes P (C layout -> A layout).
// ---------------------------------------------------------------------------
__global__ __launch_bounds__(256) void flash_attn_kernel(
        const bf16* __restrict__ Q, const bf16* __restrict__ K,
        const bf16* __restrict__ Vt, const int* __restrict__ mask,
        bf16* __restrict__ ctx) {
    int qt = blockIdx.x, h = blockIdx.y, b = blockIdx.z;
    int tid = threadIdx.x, lane = tid & 31, wave = tid >> 5;
    int q0 = qt * 128 + wave * 16;

    const bf16* Qbh = Q  + (size_t)(b * NHEAD + h) * SL * HDIM;
    const bf16* Kbh = K  + (size_t)(b * NHEAD + h) * SL * HDIM;
    const bf16* Vbh = Vt + (size_t)(b * NHEAD + h) * HDIM * SL;
    const int*  mb  = mask + b * SL;

    v16bf qf[2];
    qf[0] = load_frag(Qbh + (size_t)q0 * HDIM + 0,  HDIM, lane);
    qf[1] = load_frag(Qbh + (size_t)q0 * HDIM + 32, HDIM, lane);

    v8f O[4];
#pragma unroll
    for (int nt = 0; nt < 4; ++nt) O[nt] = (v8f)0.f;
    float mrun[8], lrun[8];
#pragma unroll
    for (int i = 0; i < 8; ++i) { mrun[i] = -INFINITY; lrun[i] = 0.f; }

    __shared__ alignas(16) bf16 Pl[8][16 * 32];   // per-wave P staging
    bf16* Pw = Pl[wave];
    int col   = lane & 15;
    int rbase = (lane & 16) ? 8 : 0;

    for (int kc = 0; kc < SL; kc += 32) {
        v8f S[2];
#pragma unroll
        for (int nt = 0; nt < 2; ++nt) {
            float mval = (float)mb[kc + nt * 16 + col] * MASK_NEG;
#pragma unroll
            for (int ii = 0; ii < 8; ++ii) S[nt][ii] = mval;
            v16bf kf0 = load_frag(Kbh + (size_t)(kc + nt * 16) * HDIM + 0,  HDIM, lane);
            v16bf kf1 = load_frag(Kbh + (size_t)(kc + nt * 16) * HDIM + 32, HDIM, lane);
            S[nt] = wmma_bf16(qf[0], kf0, S[nt]);
            S[nt] = wmma_bf16(qf[1], kf1, S[nt]);
        }
        float f[8];
#pragma unroll
        for (int ii = 0; ii < 8; ++ii) {
            float mx = fmaxf(S[0][ii], S[1][ii]);
#pragma unroll
            for (int d = 8; d >= 1; d >>= 1) mx = fmaxf(mx, __shfl_xor(mx, d, 32));
            float mnew = fmaxf(mrun[ii], mx);
            f[ii] = __expf(mrun[ii] - mnew);
            S[0][ii] = __expf(S[0][ii] - mnew);
            S[1][ii] = __expf(S[1][ii] - mnew);
            float srow = S[0][ii] + S[1][ii];
#pragma unroll
            for (int d = 8; d >= 1; d >>= 1) srow += __shfl_xor(srow, d, 32);
            lrun[ii] = lrun[ii] * f[ii] + srow;
            mrun[ii] = mnew;
        }
#pragma unroll
        for (int nt = 0; nt < 4; ++nt)
#pragma unroll
            for (int ii = 0; ii < 8; ++ii) O[nt][ii] *= f[ii];
#pragma unroll
        for (int nt = 0; nt < 2; ++nt)
#pragma unroll
            for (int ii = 0; ii < 8; ++ii)
                Pw[(rbase + ii) * 32 + nt * 16 + col] = (bf16)S[nt][ii];
        __builtin_amdgcn_wave_barrier();  // LDS is in-order per wave
        v16bf pf = load_frag(Pw, 32, lane);
#pragma unroll
        for (int nt = 0; nt < 4; ++nt) {
            v16bf vf = load_frag(Vbh + (size_t)(nt * 16) * SL + kc, SL, lane);
            O[nt] = wmma_bf16(pf, vf, O[nt]);
        }
    }

    float rinv[8];
#pragma unroll
    for (int ii = 0; ii < 8; ++ii) rinv[ii] = 1.0f / lrun[ii];
#pragma unroll
    for (int nt = 0; nt < 4; ++nt)
#pragma unroll
        for (int ii = 0; ii < 8; ++ii) {
            int row = q0 + rbase + ii;
            ctx[((size_t)(b * SL + row)) * HS + h * HDIM + nt * 16 + col] =
                (bf16)(O[nt][ii] * rinv[ii]);
        }
}

// ---------------------------------------------------------------------------
// Kernel 5: output projection + bias + residual (fp32 out).
//   out[m,n] = sum_k ctx[m,k]*Wo[n,k] + bo[n] + x[m,n]
// ---------------------------------------------------------------------------
__global__ __launch_bounds__(256) void out_gemm_kernel(
        const bf16* __restrict__ C, const bf16* __restrict__ Wo,
        const float* __restrict__ bo, const float* __restrict__ x,
        float* __restrict__ out) {
    const int m0 = blockIdx.y * 128;
    const int n0 = blockIdx.x * 128;
#if USE_TDM
    __shared__ alignas(16) bf16 As[2][128 * LDST];
    __shared__ alignas(16) bf16 Bs[2][128 * LDST];
#else
    __shared__ alignas(16) bf16 As[1][128 * LDST];
    __shared__ alignas(16) bf16 Bs[1][128 * LDST];
#endif
    int tid = threadIdx.x, lane = tid & 31, wave = tid >> 5;
    int wr = wave >> 1, wc = wave & 1;
    v8f acc[2][4];
#pragma unroll
    for (int i = 0; i < 2; ++i)
#pragma unroll
        for (int j = 0; j < 4; ++j) acc[i][j] = (v8f)0.f;

    gemm_mainloop(C, Wo, &As[0][0], &Bs[0][0], m0, n0, tid, lane, wr, wc, acc);

    int col   = lane & 15;
    int rbase = (lane & 16) ? 8 : 0;
#pragma unroll
    for (int i = 0; i < 2; ++i)
#pragma unroll
        for (int j = 0; j < 4; ++j) {
            int n = n0 + wc * 64 + j * 16 + col;
            float bval = bo[n];
#pragma unroll
            for (int ii = 0; ii < 8; ++ii) {
                int m = m0 + wr * 32 + i * 16 + rbase + ii;
                size_t idx = (size_t)m * HS + n;
                out[idx] = acc[i][j][ii] + bval + x[idx];
            }
        }
}

// ---------------------------------------------------------------------------
// Launch
// ---------------------------------------------------------------------------
extern "C" void kernel_launch(void* const* d_in, const int* in_sizes, int n_in,
                              void* d_out, int out_size, void* d_ws, size_t ws_size,
                              hipStream_t stream) {
    const float* x     = (const float*)d_in[0];   // (8,1024,1024)
    const int*   mask  = (const int*)  d_in[1];   // (8,1024)
    const float* Wqkv  = (const float*)d_in[2];   // (3,1024,1024)
    const float* bqkv  = (const float*)d_in[3];   // (3,1024)
    const float* Wo    = (const float*)d_in[4];   // (1024,1024)
    const float* bo    = (const float*)d_in[5];   // (1024,)
    const float* nw    = (const float*)d_in[6];   // (1024,)
    const float* nb    = (const float*)d_in[7];   // (1024,)
    float* out = (float*)d_out;

    char* ws = (char*)d_ws;
    bf16* Wqkv_b = (bf16*)ws;  ws += (size_t)3 * HS * HS * sizeof(bf16);
    bf16* Wo_b   = (bf16*)ws;  ws += (size_t)HS * HS * sizeof(bf16);
    bf16* Yb     = (bf16*)ws;  ws += (size_t)NROWS * HS * sizeof(bf16);
    bf16* Qb     = (bf16*)ws;  ws += (size_t)NROWS * HS * sizeof(bf16);
    bf16* Kb     = (bf16*)ws;  ws += (size_t)NROWS * HS * sizeof(bf16);
    bf16* Vtb    = (bf16*)ws;  ws += (size_t)NROWS * HS * sizeof(bf16);
    bf16* Ctxb   = (bf16*)ws;  ws += (size_t)NROWS * HS * sizeof(bf16);

    cvt_bf16_kernel<<<1024, 256, 0, stream>>>(Wqkv, Wqkv_b, 3 * HS * HS / 4);
    cvt_bf16_kernel<<<512,  256, 0, stream>>>(Wo,   Wo_b,   HS * HS / 4);
    ln_kernel<<<NROWS, 256, 0, stream>>>(x, nw, nb, Yb);
    qkv_gemm_kernel<<<dim3(24, 64), 256, 0, stream>>>(Yb, Wqkv_b, bqkv,
                                                      Qb, Kb, Vtb);
    flash_attn_kernel<<<dim3(8, NHEAD, BS), 256, 0, stream>>>(Qb, Kb, Vtb,
                                                              mask, Ctxb);
    out_gemm_kernel<<<dim3(8, 64), 256, 0, stream>>>(Ctxb, Wo_b, bo, x, out);
}
